// GNNBuildWithArchitecture_23201413333126
// MI455X (gfx1250) — compile-verified, matrix operand order checked by reading, not run on hardware
//
#include <hip/hip_runtime.h>
#include <hip/hip_bf16.h>

// ---------------------------------------------------------------------------
// GCN on MI455X (gfx1250, wave32).
// GEMMs -> v_wmma_f32_16x16x32_bf16 (bf16 in / f32 acc), LDS-staged panels.
// Aggregation -> L2-resident gather + global_atomic_add_f32 scatter.
// ---------------------------------------------------------------------------

typedef __attribute__((ext_vector_type(16))) __bf16 v16bf;
typedef __attribute__((ext_vector_type(8)))  __bf16 v8bf;
typedef __attribute__((ext_vector_type(4)))  __bf16 v4bf;
typedef __attribute__((ext_vector_type(8)))  float  v8f;

#define K_DIM   128
#define LDS_PAD 136   // 136 bf16 = 272 B row stride: 16B aligned, 64-bank spread

__device__ __forceinline__ v16bf make_frag(const __bf16* lo_p, const __bf16* hi_p) {
    v8bf lo = *(const v8bf*)lo_p;
    v8bf hi = *(const v8bf*)hi_p;
    v16bf r;
#pragma unroll
    for (int i = 0; i < 8; ++i) { r[i] = lo[i]; r[i + 8] = hi[i]; }
    return r;
}

// C[M,NO] = (RELU_IN ? relu(A) : A)[M,128] @ W[128,NO] + bias
// NT = ceil(NO/16) column tiles. Block = 256 thr = 8 waves; panel = 128 rows.
template <int NT, bool RELU_IN>
__global__ __launch_bounds__(256) void gemm_bias_kernel(
    const float* __restrict__ A, const float* __restrict__ W,
    const float* __restrict__ bias, float* __restrict__ C,
    int M, int NO)
{
    __shared__ __bf16 Ash[128][LDS_PAD];
    __shared__ __bf16 Wsh[NT * 16][LDS_PAD];

    const int tid  = threadIdx.x;
    const int row0 = blockIdx.x * 128;

    // Stage W^T (column-major per output column) as bf16, zero-pad cols >= NO.
    for (int i = tid; i < NT * 16 * K_DIM; i += 256) {
        int n = i >> 7, k = i & 127;
        float w = (n < NO) ? W[(size_t)k * NO + n] : 0.0f;
        Wsh[n][k] = (__bf16)w;
    }
    // Stage 128-row A panel as bf16 (optional fused ReLU), zero-pad rows >= M.
    for (int i = tid; i < 128 * 32; i += 256) {
        int r = i >> 5, q = (i & 31) << 2;
        int row = row0 + r;
        float4 v = make_float4(0.f, 0.f, 0.f, 0.f);
        if (row < M) v = *(const float4*)(A + (size_t)row * K_DIM + q);
        if (RELU_IN) {
            v.x = fmaxf(v.x, 0.f); v.y = fmaxf(v.y, 0.f);
            v.z = fmaxf(v.z, 0.f); v.w = fmaxf(v.w, 0.f);
        }
        v4bf b;
        b.x = (__bf16)v.x; b.y = (__bf16)v.y; b.z = (__bf16)v.z; b.w = (__bf16)v.w;
        *(v4bf*)&Ash[r][q] = b;
    }
    __syncthreads();

    const int wave  = tid >> 5;
    const int lane  = tid & 31;
    const int half  = lane >> 4;      // 0: lanes 0-15, 1: lanes 16-31
    const int ln    = lane & 15;
    const int rbase = wave * 16;      // this wave's 16-row strip in the panel

    v8f acc[NT];
#pragma unroll
    for (int nt = 0; nt < NT; ++nt) {
        int col = nt * 16 + ln;
        float bv = (bias != nullptr && col < NO) ? bias[col] : 0.0f;
#pragma unroll
        for (int r = 0; r < 8; ++r) acc[nt][r] = bv;
    }

#pragma unroll
    for (int ks = 0; ks < K_DIM; ks += 32) {
        // ISA 16-bit A/B frag layout: lanes 0-15 hold K {ks..+7, ks+16..+23},
        // lanes 16-31 hold K {ks+8..+15, ks+24..+31}.
        const int klo = ks + half * 8;
        const int khi = ks + 16 + half * 8;
        v16bf afrag = make_frag(&Ash[rbase + ln][klo], &Ash[rbase + ln][khi]);
#pragma unroll
        for (int nt = 0; nt < NT; ++nt) {
            v16bf bfrag = make_frag(&Wsh[nt * 16 + ln][klo], &Wsh[nt * 16 + ln][khi]);
            acc[nt] = __builtin_amdgcn_wmma_f32_16x16x32_bf16(
                false, afrag, false, bfrag, (short)0, acc[nt], false, false);
        }
    }

    // D layout: element r -> row half*8+r, col = lane&15 (within 16x16 tile).
#pragma unroll
    for (int nt = 0; nt < NT; ++nt) {
        int col = nt * 16 + ln;
        if (col < NO) {
#pragma unroll
            for (int r = 0; r < 8; ++r) {
                int row = row0 + rbase + half * 8 + r;
                if (row < M) C[(size_t)row * NO + col] = acc[nt][r];
            }
        }
    }
}

// ---------------- degree / normalization ----------------

__global__ __launch_bounds__(256) void deg_init_kernel(float* __restrict__ deg, int n) {
    int i = blockIdx.x * 256 + threadIdx.x;
    if (i < n) deg[i] = 1.0f;   // self-loop
}

__global__ __launch_bounds__(256) void deg_count_kernel(
    const long long* __restrict__ dst, float* __restrict__ deg, int e)
{
    int i = blockIdx.x * 256 + threadIdx.x;
    if (i < e) atomicAdd(&deg[(int)dst[i]], 1.0f);
}

__global__ __launch_bounds__(256) void dinv_kernel(
    const float* __restrict__ deg, float* __restrict__ dinv, int n)
{
    int i = blockIdx.x * 256 + threadIdx.x;
    if (i < n) dinv[i] = rsqrtf(deg[i]);
}

// agg[i][:] = bias[:] + t[i][:] * dinv[i]^2   (self-loop term, float4 per thread)
__global__ __launch_bounds__(256) void agg_init_kernel(
    const float* __restrict__ t, const float* __restrict__ bias,
    const float* __restrict__ dinv, float* __restrict__ agg, int n)
{
    int i = blockIdx.x * 256 + threadIdx.x;
    if (i >= n * 32) return;
    int node = i >> 5, q = i & 31;
    float di = dinv[node];
    float w  = di * di;
    float4 v = ((const float4*)t)[i];
    float4 b = ((const float4*)bias)[q];
    float4 o = make_float4(b.x + v.x * w, b.y + v.y * w,
                           b.z + v.z * w, b.w + v.w * w);
    ((float4*)agg)[i] = o;
}

// One wave32 per edge: lane f handles features 4f..4f+3. Coalesced 512B gather
// of t[src], scaled scatter-add into agg[dst] (L2-resident, 192MB).
__global__ __launch_bounds__(256) void edge_agg_kernel(
    const float* __restrict__ t, float* __restrict__ agg,
    const long long* __restrict__ src, const long long* __restrict__ dst,
    const float* __restrict__ dinv, int E)
{
    int lane = threadIdx.x & 31;
    int e = blockIdx.x * 8 + (threadIdx.x >> 5);
    if (e >= E) return;
    int s = (int)src[e];
    int d = (int)dst[e];
    float norm = dinv[s] * dinv[d];
    float4 v = *(const float4*)(t + (size_t)s * 128 + lane * 4);
    float* a = agg + (size_t)d * 128 + lane * 4;
    atomicAdd(a + 0, v.x * norm);
    atomicAdd(a + 1, v.y * norm);
    atomicAdd(a + 2, v.z * norm);
    atomicAdd(a + 3, v.w * norm);
}

// ---------------------------------------------------------------------------

extern "C" void kernel_launch(void* const* d_in, const int* in_sizes, int n_in,
                              void* d_out, int out_size, void* d_ws, size_t ws_size,
                              hipStream_t stream) {
    (void)in_sizes; (void)n_in; (void)out_size; (void)ws_size;

    const float*     x      = (const float*)d_in[0];
    const long long* ei     = (const long long*)d_in[1];   // int64 [2, E]
    const float*     W_pre  = (const float*)d_in[2];
    const float*     b_pre  = (const float*)d_in[3];
    const float*     W1     = (const float*)d_in[4];
    const float*     b1     = (const float*)d_in[5];
    const float*     W2     = (const float*)d_in[6];
    const float*     b2     = (const float*)d_in[7];
    const float*     W_post = (const float*)d_in[8];
    const float*     b_post = (const float*)d_in[9];
    float*           out    = (float*)d_out;

    const int N = 50000, E = 800000, D = 128, NC = 40;
    const long long* src = ei;
    const long long* dst = ei + E;

    // Workspace: 3 node-feature buffers + deg + dinv (~77 MB, L2-resident).
    char* w = (char*)d_ws;
    float* bufA = (float*)w;  w += (size_t)N * D * sizeof(float);
    float* bufB = (float*)w;  w += (size_t)N * D * sizeof(float);
    float* bufC = (float*)w;  w += (size_t)N * D * sizeof(float);
    float* deg  = (float*)w;  w += (size_t)N * sizeof(float);
    float* dinv = (float*)w;

    dim3 blk(256);
    const int gemmGrid = (N + 127) / 128;       // 391 panels of 128 rows
    const int nGrid    = (N + 255) / 256;
    const int eGrid    = (E + 255) / 256;
    const int aggGrid  = (N * 32 + 255) / 256;  // float4 units
    const int edgeGrid = (E + 7) / 8;           // 8 edges (waves) per block

    // Symmetric normalization (self-loops included)
    deg_init_kernel <<<nGrid, blk, 0, stream>>>(deg, N);
    deg_count_kernel<<<eGrid, blk, 0, stream>>>(dst, deg, E);
    dinv_kernel     <<<nGrid, blk, 0, stream>>>(deg, dinv, N);

    // h = x @ W_pre + b_pre
    gemm_bias_kernel<8, false><<<gemmGrid, blk, 0, stream>>>(x, W_pre, b_pre, bufA, N, D);

    // conv1: t = h @ W1 ; agg = b1 + D^-1/2 (A+I) D^-1/2 t ; (ReLU fused into next GEMM)
    gemm_bias_kernel<8, false><<<gemmGrid, blk, 0, stream>>>(bufA, W1, nullptr, bufB, N, D);
    agg_init_kernel <<<aggGrid, blk, 0, stream>>>(bufB, b1, dinv, bufC, N);
    edge_agg_kernel <<<edgeGrid, blk, 0, stream>>>(bufB, bufC, src, dst, dinv, E);

    // conv2: t = relu(agg1) @ W2 ; aggregate with b2
    gemm_bias_kernel<8, true><<<gemmGrid, blk, 0, stream>>>(bufC, W2, nullptr, bufA, N, D);
    agg_init_kernel <<<aggGrid, blk, 0, stream>>>(bufA, b2, dinv, bufB, N);
    edge_agg_kernel <<<edgeGrid, blk, 0, stream>>>(bufA, bufB, src, dst, dinv, E);

    // out = relu(agg2) @ W_post + b_post   (NO = 40 -> 3 column tiles, guarded)
    gemm_bias_kernel<3, true><<<gemmGrid, blk, 0, stream>>>(bufB, W_post, b_post, out, N, NC);
}
